// GCCGraphInfer_65824668778707
// MI455X (gfx1250) — matrix operand order repaired and checked by reading, not compiled, not run on hardware
//
#include <hip/hip_runtime.h>
#include <math.h>

// ---- problem constants (match reference) ----
#define NN   10000      // nodes
#define NE   160000     // edges
#define NG   64         // graphs
#define F0   128        // IN_CH
#define EDIM 16
#define OCH  16
#define H1C  256
#define H2C  128
#define EPSC 1e-5f

typedef __bf16 bf16;
typedef __attribute__((ext_vector_type(16))) __bf16 v16bf;
typedef __attribute__((ext_vector_type(8)))  __bf16 v8bf;
typedef __attribute__((ext_vector_type(8)))  float  v8f;

union AFrag { v16bf v; v8bf h[2]; };

__device__ __forceinline__ void atomicMaxF(float* a, float v) {
  if (v >= 0.f) atomicMax((int*)a, __float_as_int(v));
  else          atomicMin((unsigned int*)a, __float_as_uint(v));
}
__device__ __forceinline__ void atomicMinF(float* a, float v) {
  if (v >= 0.f) atomicMin((int*)a, __float_as_int(v));
  else          atomicMax((unsigned int*)a, __float_as_uint(v));
}

// ------------------------------------------------------------------
// tiny utility kernels
// ------------------------------------------------------------------
__global__ void k_fill(float* p, float v, long n) {
  long i = (long)blockIdx.x * blockDim.x + threadIdx.x;
  if (i < n) p[i] = v;
}
__global__ void k_cvt_bf16(const float* __restrict__ in, bf16* __restrict__ out, long n) {
  long i = (long)blockIdx.x * blockDim.x + threadIdx.x;
  if (i < n) out[i] = (bf16)in[i];
}
// edge_attr [NE,16] fp32 -> padded [NE,32] bf16 ([ea | zeros]) so the edge GEMM
// K-loop uses one uniform fragment-load pattern for every K-step.
__global__ void k_pad_edge(const float* __restrict__ ea, bf16* __restrict__ out) {
  long idx = (long)blockIdx.x * blockDim.x + threadIdx.x;
  long total = (long)NE * 32;
  if (idx >= total) return;
  long e = idx >> 5; int c = (int)(idx & 31);
  out[idx] = (c < EDIM) ? (bf16)ea[e * EDIM + c] : (bf16)0.f;
}
__global__ void k_avg_log(const int* __restrict__ hist, float* out) {
  if (threadIdx.x == 0 && blockIdx.x == 0) {
    float s = 0.f, t = 0.f;
    for (int i = 0; i < 32; ++i) { float h = (float)hist[i]; s += logf((float)i + 1.f) * h; t += h; }
    out[0] = s / t;
  }
}
__global__ void k_count(const int* __restrict__ dstA, float* __restrict__ cnt) {
  int e = blockIdx.x * blockDim.x + threadIdx.x;
  if (e < NE) atomicAdd(&cnt[dstA[e]], 1.f);
}

// Build fused edge-GEMM B matrix [2F+32, F]:
// rows [0,2F) = Wpre, rows [2F,2F+16) = We @ Wpre[2F:3F], rows [2F+16,2F+32) = 0
__global__ void k_build_bedge(const float* __restrict__ Wpre, const float* __restrict__ We,
                              int F, float* __restrict__ out) {
  long idx = (long)blockIdx.x * blockDim.x + threadIdx.x;
  long total = (long)(2 * F + 32) * F;
  if (idx >= total) return;
  int k = (int)(idx / F), o = (int)(idx % F);
  float v = 0.f;
  if (k < 2 * F) v = Wpre[(long)k * F + o];
  else if (k < 2 * F + EDIM) {
    int d = k - 2 * F; float s = 0.f;
    for (int j = 0; j < F; ++j) s += We[(long)d * F + j] * Wpre[(long)(2 * F + j) * F + o];
    v = s;
  }
  out[idx] = v;
}
// b_eff[o] = bpre[o] + sum_j be[j]*Wpre[2F+j][o]
__global__ void k_build_beff(const float* __restrict__ bpre, const float* __restrict__ be,
                             const float* __restrict__ Wpre, int F, float* __restrict__ out) {
  int o = blockIdx.x * blockDim.x + threadIdx.x;
  if (o >= F) return;
  float s = bpre[o];
  for (int j = 0; j < F; ++j) s += be[j] * Wpre[(long)(2 * F + j) * F + o];
  out[o] = s;
}

// Repack fp32 weight [K,N] (row-major) into WMMA B-operand layout (bf16):
// tile (kt,nt) -> 512 elems; lane L holds col nt*16+(L&15), K = kt*32 + 8*(L>=16) + i (i<8), kt*32 + 16 + 8*(L>=16) + (i-8) (i>=8)
__global__ void k_pack_b(const float* __restrict__ W, int Ksrc, int Kpad, int Nout,
                         bf16* __restrict__ outP) {
  long idx = (long)blockIdx.x * blockDim.x + threadIdx.x;
  long total = (long)Kpad * Nout;
  if (idx >= total) return;
  long tile = idx >> 9;
  int within = (int)(idx & 511);
  int lane = within >> 4, i = within & 15;
  int ntiles = Nout / 16;
  int kt = (int)(tile / ntiles), nt = (int)(tile % ntiles);
  int k = kt * 32 + ((lane >= 16) ? 8 : 0) + ((i < 8) ? i : 16 + (i - 8));
  int n = nt * 16 + (lane & 15);
  float v = (k < Ksrc) ? W[(long)k * Nout + n] : 0.f;
  outP[idx] = (bf16)v;
}

// ------------------------------------------------------------------
// Fused edge kernel: h = [x_dst|x_src|ea|0] @ packB + bias, scatter to node stats.
// One wave = 32 edges (2 M-tiles) x 64 cols (4 N-tiles). Double-buffered A/B.
// All A addressing via 32-bit offsets from uniform bases (SADDR form, low VGPR use).
// ------------------------------------------------------------------
__global__ void k_edge_wmma(const bf16* __restrict__ x, const bf16* __restrict__ eap,
                            const int* __restrict__ srcA, const int* __restrict__ dstA,
                            const bf16* __restrict__ packB, const float* __restrict__ bias,
                            float* __restrict__ sm, float* __restrict__ sq,
                            float* __restrict__ mnb, float* __restrict__ mxb,
                            int Fin, int Fout) {
  const int lane = threadIdx.x & 31;
  const int w    = threadIdx.x >> 5;
  const int pt   = blockIdx.x * (blockDim.x >> 5) + w;   // 32-edge pair-tile
  const int npairs = NE / 32;
  if (pt >= npairs) return;
  const int nchunk = blockIdx.y;
  const int ntiles = Fout / 16;
  const int hi = lane >> 4, mr = lane & 15;
  const int e0 = pt * 32 + mr;
  const int e1 = e0 + 16;
  // 32-bit element offsets (max ~2.6M elements, fits int)
  const int od0 = dstA[e0] * Fin + hi * 8;
  const int os0 = srcA[e0] * Fin + hi * 8;
  const int od1 = dstA[e1] * Fin + hi * 8;
  const int os1 = srcA[e1] * Fin + hi * 8;
  const int nsteps = (2 * Fin + 32) / 32;
  const bf16* bbase = packB + ((long)(nchunk * 4) << 9) + lane * 16;
  const long  bstep = (long)ntiles << 9;

  auto loadA = [&](int kt, AFrag& a0, AFrag& a1) {
    int kb = kt * 32;
    int o0, o1;
    const bf16* basep = x;
    if (kb < Fin)          { o0 = od0 + kb;        o1 = od1 + kb; }
    else if (kb < 2 * Fin) { o0 = os0 + kb - Fin;  o1 = os1 + kb - Fin; }
    else { basep = eap; o0 = e0 * 32 + hi * 8; o1 = e1 * 32 + hi * 8; } // lazy, cold path
    const bf16* p0 = basep + o0;
    const bf16* p1 = basep + o1;
    a0.h[0] = *(const v8bf*)p0; a0.h[1] = *(const v8bf*)(p0 + 16);
    a1.h[0] = *(const v8bf*)p1; a1.h[1] = *(const v8bf*)(p1 + 16);
  };
  auto loadB = [&](int kt, v16bf* b) {
    const bf16* bt = bbase + (long)kt * bstep;
#pragma unroll
    for (int nt = 0; nt < 4; ++nt) b[nt] = *(const v16bf*)(bt + (nt << 9));
  };

  v8f acc0[4], acc1[4];
  for (int t = 0; t < 4; ++t)
    for (int j = 0; j < 8; ++j) { acc0[t][j] = 0.f; acc1[t][j] = 0.f; }

  AFrag a0c, a1c; v16bf bc[4];
  loadA(0, a0c, a1c);
  loadB(0, bc);
  for (int kt = 0; kt < nsteps; ++kt) {
    const int ktn = (kt + 1 < nsteps) ? kt + 1 : kt;   // last iter: harmless reload
    AFrag a0n, a1n; v16bf bn[4];
    loadA(ktn, a0n, a1n);
    loadB(ktn, bn);
#pragma unroll
    for (int nt = 0; nt < 4; ++nt) {
      acc0[nt] = __builtin_amdgcn_wmma_f32_16x16x32_bf16(false, a0c.v, false, bc[nt],
                                                         (short)0, acc0[nt], false, false);
      acc1[nt] = __builtin_amdgcn_wmma_f32_16x16x32_bf16(false, a1c.v, false, bc[nt],
                                                         (short)0, acc1[nt], false, false);
    }
    a0c = a0n; a1c = a1n;
#pragma unroll
    for (int nt = 0; nt < 4; ++nt) bc[nt] = bn[nt];
  }

  // scatter-aggregate into node stats (L2-resident atomics), 32-bit offsets
  int off0[8], off1[8];
#pragma unroll
  for (int r = 0; r < 8; ++r) {
    off0[r] = dstA[pt * 32 + r + 8 * hi] * Fout;
    off1[r] = dstA[pt * 32 + 16 + r + 8 * hi] * Fout;
  }
#pragma unroll
  for (int nt = 0; nt < 4; ++nt) {
    int col = nchunk * 64 + nt * 16 + mr;
    float bz = bias[col];
#pragma unroll
    for (int r = 0; r < 8; ++r) {
      float v0 = acc0[nt][r] + bz;
      int   o0 = off0[r] + col;
      atomicAdd(&sm[o0], v0); atomicAdd(&sq[o0], v0 * v0);
      atomicMinF(&mnb[o0], v0); atomicMaxF(&mxb[o0], v0);
      float v1 = acc1[nt][r] + bz;
      int   o1 = off1[r] + col;
      atomicAdd(&sm[o1], v1); atomicAdd(&sq[o1], v1 * v1);
      atomicMinF(&mnb[o1], v1); atomicMaxF(&mxb[o1], v1);
    }
  }
}

// ------------------------------------------------------------------
// Generic WMMA GEMM: out[M,N](bf16) = relu?( A[M,K](bf16) @ packB + bias )
// One wave = 32 rows (2 M-tiles) x 64 cols (4 N-tiles). Double-buffered A/B.
// ------------------------------------------------------------------
__global__ void k_gemm_wmma(const bf16* __restrict__ A, int M, int K,
                            const bf16* __restrict__ packB, const float* __restrict__ bias,
                            bf16* __restrict__ out, int Nout, int relu) {
  const int lane = threadIdx.x & 31;
  const int w    = threadIdx.x >> 5;
  const int pt   = blockIdx.x * (blockDim.x >> 5) + w;   // 32-row pair-tile
  const int Mpairs = (M + 31) / 32;
  if (pt >= Mpairs) return;
  const int nchunk = blockIdx.y;
  const int ntiles = Nout / 16;
  const int hi = lane >> 4, mr = lane & 15;
  int r0 = pt * 32 + mr;      if (r0 > M - 1) r0 = M - 1;
  int r1 = pt * 32 + 16 + mr; if (r1 > M - 1) r1 = M - 1;
  const int oa0 = r0 * K + hi * 8;     // 32-bit element offsets (max ~33M, fits int)
  const int oa1 = r1 * K + hi * 8;
  const int nsteps = K / 32;
  const bf16* bbase = packB + ((long)(nchunk * 4) << 9) + lane * 16;
  const long  bstep = (long)ntiles << 9;

  auto loadA = [&](int kt, AFrag& a0, AFrag& a1) {
    const bf16* p0 = A + oa0 + kt * 32;
    const bf16* p1 = A + oa1 + kt * 32;
    a0.h[0] = *(const v8bf*)p0; a0.h[1] = *(const v8bf*)(p0 + 16);
    a1.h[0] = *(const v8bf*)p1; a1.h[1] = *(const v8bf*)(p1 + 16);
  };
  auto loadB = [&](int kt, v16bf* b) {
    const bf16* bt = bbase + (long)kt * bstep;
#pragma unroll
    for (int nt = 0; nt < 4; ++nt) b[nt] = *(const v16bf*)(bt + (nt << 9));
  };

  v8f acc0[4], acc1[4];
  for (int t = 0; t < 4; ++t)
    for (int j = 0; j < 8; ++j) { acc0[t][j] = 0.f; acc1[t][j] = 0.f; }

  AFrag a0c, a1c; v16bf bc[4];
  loadA(0, a0c, a1c);
  loadB(0, bc);
  for (int kt = 0; kt < nsteps; ++kt) {
    const int ktn = (kt + 1 < nsteps) ? kt + 1 : kt;
    AFrag a0n, a1n; v16bf bn[4];
    loadA(ktn, a0n, a1n);
    loadB(ktn, bn);
#pragma unroll
    for (int nt = 0; nt < 4; ++nt) {
      acc0[nt] = __builtin_amdgcn_wmma_f32_16x16x32_bf16(false, a0c.v, false, bc[nt],
                                                         (short)0, acc0[nt], false, false);
      acc1[nt] = __builtin_amdgcn_wmma_f32_16x16x32_bf16(false, a1c.v, false, bc[nt],
                                                         (short)0, acc1[nt], false, false);
    }
    a0c = a0n; a1c = a1n;
#pragma unroll
    for (int nt = 0; nt < 4; ++nt) bc[nt] = bn[nt];
  }

#pragma unroll
  for (int nt = 0; nt < 4; ++nt) {
    int col = nchunk * 64 + nt * 16 + mr;
    float bz = bias[col];
#pragma unroll
    for (int r = 0; r < 8; ++r) {
      int o0 = pt * 32 + r + 8 * hi;
      if (o0 < M) {
        float v = acc0[nt][r] + bz;
        if (relu) v = fmaxf(v, 0.f);
        out[o0 * Nout + col] = (bf16)v;
      }
      int o1 = pt * 32 + 16 + r + 8 * hi;
      if (o1 < M) {
        float v = acc1[nt][r] + bz;
        if (relu) v = fmaxf(v, 0.f);
        out[o1 * Nout + col] = (bf16)v;
      }
    }
  }
}

// ------------------------------------------------------------------
// Build the 13F concat feature matrix [N, 13F] in bf16 from aggregates.
// layout: [x | mean,min,max,std | (..)*s1 | (..)*s2]
// ------------------------------------------------------------------
__global__ void k_build_concat(const bf16* __restrict__ x, int F,
                               const float* __restrict__ sm, const float* __restrict__ sq,
                               const float* __restrict__ mnb, const float* __restrict__ mxb,
                               const float* __restrict__ cnt, const float* __restrict__ avgp,
                               bf16* __restrict__ Acat) {
  long idx = (long)blockIdx.x * blockDim.x + threadIdx.x;
  long total = (long)NN * F;
  if (idx >= total) return;
  int n = (int)(idx / F), c = (int)(idx % F);
  float ct = cnt[n];
  float deg = fmaxf(ct, 1.f);
  long o = (long)n * F + c;
  float mean = sm[o] / deg;
  float m2   = sq[o] / deg;
  float var  = m2 - mean * mean; if (var < 0.f) var = 0.f;
  float sd   = sqrtf(var + EPSC);
  float vmn = (ct > 0.f) ? mnb[o] : 0.f;
  float vmx = (ct > 0.f) ? mxb[o] : 0.f;
  float al = avgp[0];
  float ld = logf(deg + 1.f);
  float s1 = ld / al, s2 = al / ld;
  long Kc = 13L * F;
  bf16* row = Acat + (long)n * Kc;
  row[c] = x[o];
  float ag[4] = {mean, vmn, vmx, sd};
#pragma unroll
  for (int j = 0; j < 4; ++j) {
    row[F + j * F + c]      = (bf16)ag[j];
    row[5 * F + j * F + c]  = (bf16)(ag[j] * s1);
    row[9 * F + j * F + c]  = (bf16)(ag[j] * s2);
  }
}

// ------------------------------------------------------------------
// global mean pool (atomic) + head (dense -> BN -> leaky -> dense)
// ------------------------------------------------------------------
__global__ void k_pool(const bf16* __restrict__ x2, const int* __restrict__ batch,
                       float* __restrict__ gsum, float* __restrict__ gcnt) {
  long idx = (long)blockIdx.x * blockDim.x + threadIdx.x;
  long total = (long)NN * H2C;
  if (idx >= total) return;
  int n = (int)(idx / H2C), c = (int)(idx % H2C);
  int b = batch[n];
  atomicAdd(&gsum[(long)b * H2C + c], (float)x2[idx]);
  if (c == 0) atomicAdd(&gcnt[b], 1.f);
}

__global__ void k_head(const float* __restrict__ gsum, const float* __restrict__ gcnt,
                       const float* __restrict__ Wd1, const float* __restrict__ bd1,
                       const float* __restrict__ gamma, const float* __restrict__ beta,
                       const float* __restrict__ Wd2, const float* __restrict__ bd2,
                       float* __restrict__ out) {
  __shared__ float y[NG * 64];
  __shared__ float mu[64], iv[64];
  int t = threadIdx.x;                      // 256 threads
  for (int p = t; p < NG * 64; p += 256) {
    int g = p / 64, o = p % 64;
    float ic = 1.f / fmaxf(gcnt[g], 1.f);
    float s = bd1[o];
    for (int c = 0; c < H2C; ++c) s += gsum[(long)g * H2C + c] * ic * Wd1[(long)c * 64 + o];
    y[p] = s;
  }
  __syncthreads();
  if (t < 64) {
    float m = 0.f;
    for (int g = 0; g < NG; ++g) m += y[g * 64 + t];
    m /= (float)NG;
    float v = 0.f;
    for (int g = 0; g < NG; ++g) { float d = y[g * 64 + t] - m; v += d * d; }
    v /= (float)NG;
    mu[t] = m; iv[t] = rsqrtf(v + EPSC);
  }
  __syncthreads();
  for (int p = t; p < NG * OCH; p += 256) {
    int g = p / OCH, o = p % OCH;
    float s = bd2[o];
    for (int c = 0; c < 64; ++c) {
      float yy = (y[g * 64 + c] - mu[c]) * iv[c] * gamma[c] + beta[c];
      yy = (yy > 0.f) ? yy : 0.1f * yy;
      s += yy * Wd2[c * OCH + o];
    }
    out[p] = s;
  }
}

// ------------------------------------------------------------------
#define CDIV(a, b) (((a) + (b) - 1) / (b))

extern "C" void kernel_launch(void* const* d_in, const int* in_sizes, int n_in,
                              void* d_out, int out_size, void* d_ws, size_t ws_size,
                              hipStream_t stream) {
  const float* node_attr = (const float*)d_in[0];
  const int*   eidx      = (const int*)d_in[1];
  const int*   srcA = eidx;            // edge_index[0]
  const int*   dstA = eidx + NE;       // edge_index[1]
  const float* edge_attr = (const float*)d_in[2];
  const int*   batch     = (const int*)d_in[3];
  const int*   deg_hist  = (const int*)d_in[4];
  const float *We1=(const float*)d_in[5],  *be1=(const float*)d_in[6];
  const float *Wpre1=(const float*)d_in[7],*bpre1=(const float*)d_in[8];
  const float *Wpost1=(const float*)d_in[9],*bpost1=(const float*)d_in[10];
  const float *Wlin1=(const float*)d_in[11],*blin1=(const float*)d_in[12];
  const float *We2=(const float*)d_in[13], *be2=(const float*)d_in[14];
  const float *Wpre2=(const float*)d_in[15],*bpre2=(const float*)d_in[16];
  const float *Wpost2=(const float*)d_in[17],*bpost2=(const float*)d_in[18];
  const float *Wlin2=(const float*)d_in[19],*blin2=(const float*)d_in[20];
  const float *Wd1=(const float*)d_in[21], *bd1=(const float*)d_in[22];
  const float *gamma=(const float*)d_in[23],*beta=(const float*)d_in[24];
  const float *Wd2=(const float*)d_in[25], *bd2=(const float*)d_in[26];

  // ---- workspace bump allocator (256B aligned) ----
  char* base = (char*)d_ws;
  size_t off = 0;
  auto alloc = [&](size_t bytes) -> void* {
    void* p = base + off;
    off += (bytes + 255) & ~(size_t)255;
    return p;
  };
  float* avgp  = (float*)alloc(256);
  bf16*  x0    = (bf16*)alloc((size_t)NN * F0 * 2);
  bf16*  eapad = (bf16*)alloc((size_t)NE * 32 * 2);
  bf16*  x1    = (bf16*)alloc((size_t)NN * H1C * 2);
  bf16*  x2    = (bf16*)alloc((size_t)NN * H2C * 2);
  bf16*  t1    = (bf16*)alloc((size_t)NN * H1C * 2);
  bf16*  t2    = (bf16*)alloc((size_t)NN * H2C * 2);
  bf16*  Acat  = (bf16*)alloc((size_t)NN * 13 * H1C * 2);        // sized for layer 2 (13*256)
  float* sm    = (float*)alloc((size_t)NN * H1C * 4);
  float* sq    = (float*)alloc((size_t)NN * H1C * 4);
  float* mnb   = (float*)alloc((size_t)NN * H1C * 4);
  float* mxb   = (float*)alloc((size_t)NN * H1C * 4);
  float* cnt   = (float*)alloc((size_t)NN * 4);
  float* btmp  = (float*)alloc((size_t)(2 * H1C + 32) * H1C * 4); // fp32 fused-B scratch
  bf16*  pPre1 = (bf16*)alloc((size_t)288 * F0 * 2);
  bf16*  pPre2 = (bf16*)alloc((size_t)544 * H1C * 2);
  bf16*  pPost1= (bf16*)alloc((size_t)(13 * F0) * H1C * 2);
  bf16*  pPost2= (bf16*)alloc((size_t)(13 * H1C) * H2C * 2);
  bf16*  pLin1 = (bf16*)alloc((size_t)H1C * H1C * 2);
  bf16*  pLin2 = (bf16*)alloc((size_t)H2C * H2C * 2);
  float* beff1 = (float*)alloc(F0 * 4);
  float* beff2 = (float*)alloc(H1C * 4);
  float* gsum  = (float*)alloc((size_t)NG * H2C * 4);
  float* gcnt  = (float*)alloc((size_t)NG * 4);
  (void)ws_size; (void)in_sizes; (void)n_in; (void)out_size;

  const int Kp1 = 2 * F0 + 32;   // 288
  const int Kp2 = 2 * H1C + 32;  // 544

  // ---- scalars + bf16 conversions ----
  k_avg_log<<<1, 32, 0, stream>>>(deg_hist, avgp);
  { long n = (long)NN * F0;  k_cvt_bf16<<<CDIV(n,256),256,0,stream>>>(node_attr, x0, n); }
  { long n = (long)NE * 32;  k_pad_edge<<<CDIV(n,256),256,0,stream>>>(edge_attr, eapad); }

  // ---- weight prep: fuse edge-encoder into pre_nn, repack all B operands ----
  { long n = (long)Kp1 * F0;  k_build_bedge<<<CDIV(n,256),256,0,stream>>>(Wpre1, We1, F0, btmp);
                              k_pack_b<<<CDIV(n,256),256,0,stream>>>(btmp, Kp1, Kp1, F0, pPre1); }
  k_build_beff<<<CDIV(F0,128),128,0,stream>>>(bpre1, be1, Wpre1, F0, beff1);
  { long n = (long)Kp2 * H1C; k_build_bedge<<<CDIV(n,256),256,0,stream>>>(Wpre2, We2, H1C, btmp);
                              k_pack_b<<<CDIV(n,256),256,0,stream>>>(btmp, Kp2, Kp2, H1C, pPre2); }
  k_build_beff<<<CDIV(H1C,128),128,0,stream>>>(bpre2, be2, Wpre2, H1C, beff2);
  { long n = (long)(13*F0)*H1C;  k_pack_b<<<CDIV(n,256),256,0,stream>>>(Wpost1, 13*F0, 13*F0, H1C, pPost1); }
  { long n = (long)H1C*H1C;      k_pack_b<<<CDIV(n,256),256,0,stream>>>(Wlin1,  H1C,   H1C,   H1C, pLin1); }
  { long n = (long)(13*H1C)*H2C; k_pack_b<<<CDIV(n,256),256,0,stream>>>(Wpost2, 13*H1C,13*H1C,H2C, pPost2); }
  { long n = (long)H2C*H2C;      k_pack_b<<<CDIV(n,256),256,0,stream>>>(Wlin2,  H2C,   H2C,   H2C, pLin2); }

  // ---- in-degree (shared by both layers) ----
  k_fill<<<CDIV((long)NN,256),256,0,stream>>>(cnt, 0.f, NN);
  k_count<<<CDIV(NE,256),256,0,stream>>>(dstA, cnt);

  // ================= PNA layer 1 (F=128) =================
  { long n = (long)NN * F0;
    k_fill<<<CDIV(n,256),256,0,stream>>>(sm, 0.f, n);
    k_fill<<<CDIV(n,256),256,0,stream>>>(sq, 0.f, n);
    k_fill<<<CDIV(n,256),256,0,stream>>>(mnb,  INFINITY, n);
    k_fill<<<CDIV(n,256),256,0,stream>>>(mxb, -INFINITY, n);
  }
  k_edge_wmma<<<dim3(NE/32/8, F0/64), 256, 0, stream>>>(
      x0, eapad, srcA, dstA, pPre1, beff1, sm, sq, mnb, mxb, F0, F0);
  { long n = (long)NN * F0;
    k_build_concat<<<CDIV(n,256),256,0,stream>>>(x0, F0, sm, sq, mnb, mxb, cnt, avgp, Acat); }
  k_gemm_wmma<<<dim3(CDIV(CDIV(NN,32),8), H1C/64), 256, 0, stream>>>(
      Acat, NN, 13*F0, pPost1, bpost1, t1, H1C, 0);
  k_gemm_wmma<<<dim3(CDIV(CDIV(NN,32),8), H1C/64), 256, 0, stream>>>(
      t1, NN, H1C, pLin1, blin1, x1, H1C, 1);

  // ================= PNA layer 2 (F=256) =================
  { long n = (long)NN * H1C;
    k_fill<<<CDIV(n,256),256,0,stream>>>(sm, 0.f, n);
    k_fill<<<CDIV(n,256),256,0,stream>>>(sq, 0.f, n);
    k_fill<<<CDIV(n,256),256,0,stream>>>(mnb,  INFINITY, n);
    k_fill<<<CDIV(n,256),256,0,stream>>>(mxb, -INFINITY, n);
  }
  k_edge_wmma<<<dim3(NE/32/8, H1C/64), 256, 0, stream>>>(
      x1, eapad, srcA, dstA, pPre2, beff2, sm, sq, mnb, mxb, H1C, H1C);
  { long n = (long)NN * H1C;
    k_build_concat<<<CDIV(n,256),256,0,stream>>>(x1, H1C, sm, sq, mnb, mxb, cnt, avgp, Acat); }
  k_gemm_wmma<<<dim3(CDIV(CDIV(NN,32),8), H2C/64), 256, 0, stream>>>(
      Acat, NN, 13*H1C, pPost2, bpost2, t2, H2C, 0);
  k_gemm_wmma<<<dim3(CDIV(CDIV(NN,32),8), H2C/64), 256, 0, stream>>>(
      t2, NN, H2C, pLin2, blin2, x2, H2C, 1);

  // ================= pool + head =================
  k_fill<<<CDIV((long)NG*H2C,256),256,0,stream>>>(gsum, 0.f, (long)NG*H2C);
  k_fill<<<1,64,0,stream>>>(gcnt, 0.f, NG);
  { long n = (long)NN * H2C; k_pool<<<CDIV(n,256),256,0,stream>>>(x2, batch, gsum, gcnt); }
  k_head<<<1, 256, 0, stream>>>(gsum, gcnt, Wd1, bd1, gamma, beta, Wd2, bd2, (float*)d_out);
}
